// BiLinearDotLayer_72378788872869
// MI455X (gfx1250) — compile-verified, matrix operand order checked by reading, not run on hardware
//
#include <hip/hip_runtime.h>
#include <hip/hip_bf16.h>

// BiLinearDotLayer for MI455X (gfx1250, wave32, WMMA).
//   Stage 1 (batch-major GEMM): bil[b,i,:] = feat[b,i,:] @ W_i   -> bf16 in d_ws
//   Stage 2 (per-batch GEMM)  : out[b,i,j] = bil[b,i,:] . feat[b,j,:]
// Both stages compute TRANSPOSED 16x16 tiles (operands swapped) so each lane's
// 8 accumulators are contiguous along the fast output axis -> vector stores.
// All LDS fragment reads are 2x ds_load_b128; stage-2 bilinear operand is
// DMA'd global->LDS with global_load_async_to_lds_b128 (ASYNCcnt path).
// d_ws requirement: B*F*E*2 bytes = 52,428,800 bytes of scratch.

#define BDIM 2048
#define FDIM 200
#define EDIM 64
#define FPAD 208          // 13 * 16, padded row count for stage-2 LDS tiles
#define NTILE 13          // ceil(200/16)
#define ROWP 36           // padded LDS row stride (dwords): 144B, 16B-aligned

typedef __bf16 bf16_t;
typedef bf16_t v16bf __attribute__((ext_vector_type(16)));
typedef float  v8f   __attribute__((ext_vector_type(8)));
typedef float  v4f   __attribute__((ext_vector_type(4)));   // for NT stores

union ABFrag { v16bf v; unsigned int u[8]; uint4 q[2]; };

// round-to-nearest-even f32 -> bf16, packed pair
__device__ __forceinline__ unsigned int pack_bf16(float x, float y) {
  unsigned int xu = __float_as_uint(x);
  unsigned int yu = __float_as_uint(y);
  xu += 0x7FFFu + ((xu >> 16) & 1u);
  yu += 0x7FFFu + ((yu >> 16) & 1u);
  return (xu >> 16) | (yu & 0xFFFF0000u);
}

// CDNA5 async global->LDS copy (16B per lane), tracked by ASYNCcnt.
// Flat LDS addresses carry the DS offset in the low 32 bits (aperture above).
__device__ __forceinline__ void async_copy_b128(void* lds_ptr, const void* gptr) {
  unsigned int lds_off = (unsigned int)(size_t)lds_ptr;
  unsigned long long ga = (unsigned long long)gptr;
  asm volatile("global_load_async_to_lds_b128 %0, %1, off"
               :
               : "v"(lds_off), "v"(ga)
               : "memory");
}
__device__ __forceinline__ void wait_asynccnt0() {
  asm volatile("s_wait_asynccnt 0x0" ::: "memory");
}

// 16-bit A/B fragment (ISA 7.12.2): lanes 0-15 hold K={0..7,16..23}, lanes
// 16-31 hold K={8..15,24..31}; dwords pack consecutive K pairs. With rows of
// packed pairs in LDS, a fragment is two contiguous uint4s per lane:
// kp = 16h + 4*half16 + {0..3} and +8.
__device__ __forceinline__ ABFrag load_frag_b128(const unsigned int* row,
                                                 int half16, int h) {
  ABFrag f;
  int base = h * 16 + half16 * 4;
  f.q[0] = *(const uint4*)(row + base);
  f.q[1] = *(const uint4*)(row + base + 8);
  return f;
}

__device__ __forceinline__ v8f wmma2(const ABFrag& a0, const ABFrag& b0,
                                     const ABFrag& a1, const ABFrag& b1) {
  v8f c = {0.f, 0.f, 0.f, 0.f, 0.f, 0.f, 0.f, 0.f};
  c = __builtin_amdgcn_wmma_f32_16x16x32_bf16(false, a0.v, false, b0.v,
                                              (short)0, c, false, false);
  c = __builtin_amdgcn_wmma_f32_16x16x32_bf16(false, a1.v, false, b1.v,
                                              (short)0, c, false, false);
  return c;
}

// Two float4 non-temporal stores of the 8 accumulators (contiguous fast axis).
__device__ __forceinline__ void store_tile_nt(float* obase, const v8f& c) {
  v4f lo = {c[0], c[1], c[2], c[3]};
  v4f hi = {c[4], c[5], c[6], c[7]};
  __builtin_nontemporal_store(lo, (v4f*)obase);
  __builtin_nontemporal_store(hi, (v4f*)(obase + 4));
}

// ---------------------------------------------------------------------------
// Stage 1: grid (FDIM, BDIM/64), block 128 (4 waves). Wave w handles batches
// [b0, b0+16) for feature index i = blockIdx.x. Computes the TRANSPOSED tile
//   D (d x m) = W_i^T (64x64) @ feat^T (64x16)
// so lane (half16, mn) holds d = t*16 + 8*half16 + 0..7 for batch b0+mn:
// 4 packed dwords -> one b128 store of 8 consecutive bf16 per tile.
// ---------------------------------------------------------------------------
__global__ __launch_bounds__(128) void bilinear_stage1(
    const float* __restrict__ feat, const float* __restrict__ weight,
    unsigned short* __restrict__ bil) {
  __shared__ unsigned int wCol[EDIM * ROWP];     // [d][e/2]  pairs along e
  __shared__ unsigned int aLds[4][16 * ROWP];    // per-wave [m][e/2]

  const int i    = blockIdx.x;
  const int tid  = threadIdx.x;
  const int wave = tid >> 5;
  const int lane = tid & 31;
  const int b0   = blockIdx.y * 64 + wave * 16;

  // W_i column-major fill: entry (d, q) packs w[8q+0..7][d] (coalesced in d)
  const float* wbase = weight + (size_t)i * EDIM * EDIM;
  for (int idx = tid; idx < EDIM * 8; idx += 128) {
    int q = idx >> 6, d = idx & 63;
    float x0 = wbase[(8 * q + 0) * EDIM + d], x1 = wbase[(8 * q + 1) * EDIM + d];
    float x2 = wbase[(8 * q + 2) * EDIM + d], x3 = wbase[(8 * q + 3) * EDIM + d];
    float x4 = wbase[(8 * q + 4) * EDIM + d], x5 = wbase[(8 * q + 5) * EDIM + d];
    float x6 = wbase[(8 * q + 6) * EDIM + d], x7 = wbase[(8 * q + 7) * EDIM + d];
    uint4 pk = {pack_bf16(x0, x1), pack_bf16(x2, x3),
                pack_bf16(x4, x5), pack_bf16(x6, x7)};
    *(uint4*)&wCol[d * ROWP + q * 4] = pk;
  }
  // per-wave feat tile: rows are e-contiguous -> float4 loads, pack pairs
  for (int idx = lane; idx < 16 * 8; idx += 32) {
    int m = idx >> 3, q = idx & 7;
    const float4* fb =
        (const float4*)(feat + ((size_t)(b0 + m) * FDIM + i) * EDIM + q * 8);
    float4 x = fb[0], y = fb[1];
    uint4 pk = {pack_bf16(x.x, x.y), pack_bf16(x.z, x.w),
                pack_bf16(y.x, y.y), pack_bf16(y.z, y.w)};
    *(uint4*)&aLds[wave][m * ROWP + q * 4] = pk;
  }
  __syncthreads();

  const int half16 = lane >> 4;
  const int mn     = lane & 15;

  // B operand (feat^T): lane = batch column, pairs along e
  ABFrag bf0 = load_frag_b128(&aLds[wave][mn * ROWP], half16, 0);
  ABFrag bf1 = load_frag_b128(&aLds[wave][mn * ROWP], half16, 1);

#pragma unroll
  for (int t = 0; t < 4; ++t) {                  // d-tiles of 16
    const unsigned int* wrow = &wCol[(t * 16 + mn) * ROWP];
    ABFrag w0 = load_frag_b128(wrow, half16, 0); // A operand: W^T rows (d)
    ABFrag w1 = load_frag_b128(wrow, half16, 1);

    v8f c = wmma2(w0, bf0, w1, bf1);

    int d0 = t * 16 + half16 * 8;
    uint4 pk = {pack_bf16(c[0], c[1]), pack_bf16(c[2], c[3]),
                pack_bf16(c[4], c[5]), pack_bf16(c[6], c[7])};
    // RT store: bilinear (52 MB) stays resident in the 192 MB L2 for stage 2
    *(uint4*)(bil + ((size_t)(b0 + mn) * FDIM + i) * EDIM + d0) = pk;
  }
}

// ---------------------------------------------------------------------------
// Stage 2: grid (BDIM), block 256 (8 waves). One batch per workgroup.
// Computes the TRANSPOSED tile  D' (j x i) = feat (16x64) @ bil^T (64x16)
// so lane (half16, mn) holds j = j0 + 8*half16 + 0..7 for row i0+mn:
// two float4 non-temporal stores per tile. Interior (144) tiles store
// unguarded; 25 edge tiles use whole-vector guards (200-192 == 8).
// ---------------------------------------------------------------------------
__global__ __launch_bounds__(256) void bilinear_stage2(
    const float* __restrict__ feat, const unsigned int* __restrict__ bil,
    float* __restrict__ out) {
  __shared__ unsigned int fLds[FPAD * ROWP];   // [j][d/2] bf16 pairs of feat
  __shared__ unsigned int pLds[FPAD * ROWP];   // [i][d/2] bf16 pairs of bilinear

  const int b    = blockIdx.x;
  const int tid  = threadIdx.x;
  const int wave = tid >> 5;
  const int lane = tid & 31;

  for (int idx = tid; idx < FPAD * 8; idx += 256) {   // 16B chunks
    int j = idx >> 3, q = idx & 7;
    if (j < FDIM) {
      // bilinear: already packed bf16 in ws -> async DMA straight into LDS
      async_copy_b128(&pLds[j * ROWP + q * 4],
                      bil + (size_t)b * FDIM * 32 + j * 32 + q * 4);
      // feature: load f32, convert+pack to bf16 pairs along d
      const float4* fb =
          (const float4*)(feat + ((size_t)b * FDIM + j) * EDIM + q * 8);
      float4 x = fb[0], y = fb[1];
      uint4 pk = {pack_bf16(x.x, x.y), pack_bf16(x.z, x.w),
                  pack_bf16(y.x, y.y), pack_bf16(y.z, y.w)};
      *(uint4*)&fLds[j * ROWP + q * 4] = pk;
    } else {
      uint4 z = {0u, 0u, 0u, 0u};
      *(uint4*)&pLds[j * ROWP + q * 4] = z;
      *(uint4*)&fLds[j * ROWP + q * 4] = z;
    }
  }
  wait_asynccnt0();          // drain async global->LDS before the barrier
  __syncthreads();

  const int half16 = lane >> 4;
  const int mn     = lane & 15;

  // interior tiles: ti, tj in 0..11 -> unguarded vector stores
  for (int tt = wave; tt < 144; tt += 8) {             // wave-uniform
    int i0 = (tt / 12) * 16;
    int j0 = (tt % 12) * 16;

    ABFrag a0 = load_frag_b128(&fLds[(j0 + mn) * ROWP], half16, 0);
    ABFrag a1 = load_frag_b128(&fLds[(j0 + mn) * ROWP], half16, 1);
    ABFrag b0 = load_frag_b128(&pLds[(i0 + mn) * ROWP], half16, 0);
    ABFrag b1 = load_frag_b128(&pLds[(i0 + mn) * ROWP], half16, 1);
    v8f c = wmma2(a0, b0, a1, b1);

    store_tile_nt(out + ((size_t)b * FDIM + i0 + mn) * FDIM + j0 + half16 * 8,
                  c);
  }

  // edge tiles: (12, 0..12) and (0..11, 12)
  for (int e = wave; e < 2 * NTILE - 1; e += 8) {      // 25 tiles, wave-uniform
    int ti = (e < NTILE) ? 12 : (e - NTILE);
    int tj = (e < NTILE) ? e : 12;
    int i0 = ti * 16, j0 = tj * 16;

    ABFrag a0 = load_frag_b128(&fLds[(j0 + mn) * ROWP], half16, 0);
    ABFrag a1 = load_frag_b128(&fLds[(j0 + mn) * ROWP], half16, 1);
    ABFrag b0 = load_frag_b128(&pLds[(i0 + mn) * ROWP], half16, 0);
    ABFrag b1 = load_frag_b128(&pLds[(i0 + mn) * ROWP], half16, 1);
    v8f c = wmma2(a0, b0, a1, b1);

    int icol = i0 + mn;
    int jrow = j0 + half16 * 8;
    if (icol < FDIM && (jrow + 7) < FDIM) {   // 8-runs align with the 200 edge
      store_tile_nt(out + ((size_t)b * FDIM + icol) * FDIM + jrow, c);
    }
  }
}

extern "C" void kernel_launch(void* const* d_in, const int* in_sizes, int n_in,
                              void* d_out, int out_size, void* d_ws, size_t ws_size,
                              hipStream_t stream) {
  const float* feature = (const float*)d_in[0];   // (2048, 200, 64) f32
  const float* weight  = (const float*)d_in[1];   // (200, 64, 64)  f32
  float* out = (float*)d_out;                     // (2048, 200, 200) f32

  // bilinear scratch: B*F*E bf16 = 52.4 MB in d_ws
  unsigned short* bil_w = (unsigned short*)d_ws;
  const unsigned int* bil_r = (const unsigned int*)d_ws;

  bilinear_stage1<<<dim3(FDIM, BDIM / 64), 128, 0, stream>>>(feature, weight, bil_w);
  bilinear_stage2<<<dim3(BDIM), 256, 0, stream>>>(feature, bil_r, out);
}
